// NeuralPDA_86328842649766
// MI455X (gfx1250) — compile-verified, matrix operand order checked by reading, not run on hardware
//
#include <hip/hip_runtime.h>
#include <hip/hip_bf16.h>

typedef __bf16 bf16;
typedef __attribute__((ext_vector_type(16))) __bf16 v16bf;
typedef __attribute__((ext_vector_type(8)))  __bf16 v8bf;
typedef __attribute__((ext_vector_type(8)))  float  v8f;

#define DEV static __device__ __forceinline__

// dims
#define B_   32
#define L_   128
#define V_   32000
#define D_   256
#define NT_  512
#define CD_  128
#define DEPTH_ 130

// d_out offsets (float elements)
#define OUT_LOGITS 0L
#define OUT_PUSH   131072000L
#define OUT_CODES  131080192L
#define OUT_M      132128768L
#define OUT_N      132194304L

// workspace offsets (bytes)
#define WS_HBF   0L            // bf16 [4096][256]  (row = t*32 + b)
#define WS_WOUTS 2097152L      // bf16 swizzled Wout frags (16,384,000 B)
#define WS_WI    18481152L     // bf16 swizzled 256x256 (131072 B)
#define WS_WH    18612224L
#define WS_WS    18743296L     // 128x256 (65536 B)
#define WS_WT    18808832L
#define WS_WN    18939904L
#define WS_WC    19070976L
#define WS_CBT   19202048L     // codebook^T swizzled 128x512 (131072 B)
#define WS_AM    19333120L     // f32 [512][128]
#define WS_AN    19595264L     // f32 [512]
#define WS_STACK 19597312L     // f32 [32][130][128]

// ---------------------------------------------------------------------------
// WMMA + fragment helpers (V_WMMA_F32_16X16X32_BF16, wave32)
// ---------------------------------------------------------------------------
DEV v8f vzero8() {
  v8f z;
#pragma unroll
  for (int i = 0; i < 8; ++i) z[i] = 0.f;
  return z;
}

DEV v8f wmma_bf16(v16bf a, v16bf b, v8f c) {
  return __builtin_amdgcn_wmma_f32_16x16x32_bf16(false, a, false, b,
                                                 (short)0, c, false, false);
}

DEV v16bf cat8(v8bf lo, v8bf hi) {
  return __builtin_shufflevector(lo, hi, 0, 1, 2, 3, 4, 5, 6, 7,
                                 8, 9, 10, 11, 12, 13, 14, 15);
}

// A 16x32 bf16 fragment from row-major bf16 LDS: per ISA layout each lane's
// 16 elements are two contiguous 16B runs -> two ds_load_b128.
DEV v16bf load_a_ldsbf(const bf16* A, int lda, int m0, int k0, int lane) {
  int m = m0 + (lane & 15), hi = (lane >> 4) & 1;
  const bf16* base = A + m * lda + k0 + hi * 8;
  return cat8(*(const v8bf*)(base), *(const v8bf*)(base + 16));
}

// A fragment from row-major bf16 global (two global_load_b128)
DEV v16bf load_a_gbf(const bf16* A, int lda, long m0, int k0, int lane) {
  long m = m0 + (lane & 15); int hi = (lane >> 4) & 1;
  const bf16* base = A + m * lda + k0 + hi * 8;
  return cat8(*(const v8bf*)(base), *(const v8bf*)(base + 16));
}

// B fragment from pre-swizzled stream: fragment f = (n0/16)*kfr + (k0/32),
// 512 bf16 per fragment, lane-major -> each lane reads 32 contiguous bytes.
DEV v16bf load_b_swz(const bf16* Wsz, int kfr, int k0, int n0, int lane) {
  long f = (long)(n0 >> 4) * kfr + (k0 >> 5);
  const bf16* p = Wsz + (f << 9) + lane * 16;
  return cat8(*(const v8bf*)(p), *(const v8bf*)(p + 8));
}

// shared 32xN = A(32xK,LDS bf16) @ Wsz(KxN swizzled); wave owns n0..n0+31
DEV void gemm_strip(const bf16* Alds, int lda, int K, const bf16* Wsz,
                    int n0, int lane, v8f acc[2][2]) {
  const int kfr = K >> 5;
  for (int k0 = 0; k0 < K; k0 += 32) {
    v16bf a0 = load_a_ldsbf(Alds, lda, 0, k0, lane);
    v16bf a1 = load_a_ldsbf(Alds, lda, 16, k0, lane);
#pragma unroll
    for (int nt = 0; nt < 2; ++nt) {
      v16bf bb = load_b_swz(Wsz, kfr, k0, n0 + nt * 16, lane);
      acc[0][nt] = wmma_bf16(a0, bb, acc[0][nt]);
      acc[1][nt] = wmma_bf16(a1, bb, acc[1][nt]);
    }
  }
}

// ---------------------------------------------------------------------------
// K0: zero workspace accumulators + stack buffer (harness does not re-poison)
// ---------------------------------------------------------------------------
__global__ void k_zero_ws(float* amAcc, float* anAcc, float* stk) {
  long i = (long)blockIdx.x * blockDim.x + threadIdx.x;
  const long NA = NT_ * CD_, NN = NT_, NS = (long)B_ * DEPTH_ * CD_;
  if (i < NA) amAcc[i] = 0.f;
  else if (i < NA + NN) anAcc[i - NA] = 0.f;
  else if (i < NA + NN + NS) stk[i - NA - NN] = 0.f;
}

// ---------------------------------------------------------------------------
// K1: swizzle a KxN f32 weight (optionally source-transposed: W[n][k]) into
// the fragment-native bf16 layout. One thread per output element.
// ---------------------------------------------------------------------------
__global__ void k_swizzle(const float* __restrict__ W, bf16* __restrict__ dst,
                          int K, int N, int src_ld, int transposed) {
  long i = (long)blockIdx.x * blockDim.x + threadIdx.x;
  if (i >= (long)K * N) return;
  long frag = i >> 9;
  int  rem  = (int)(i & 511);
  int  lane = rem >> 4, e = rem & 15;
  int  kfr  = K >> 5;
  int  n0   = (int)(frag / kfr) * 16;
  int  k0   = (int)(frag % kfr) * 32;
  int  n  = n0 + (lane & 15);
  int  hi = (lane >> 4) & 1;
  int  k  = k0 + hi * 8 + ((e < 8) ? e : (e + 8));
  float v = transposed ? W[(long)n * src_ld + k] : W[(long)k * src_ld + n];
  dst[i] = (bf16)v;
}

// ---------------------------------------------------------------------------
// K2: sequential PDA recurrence — 1 block, 8 wave32 waves, all matmuls WMMA
// ---------------------------------------------------------------------------
__global__ void __launch_bounds__(256)
k_recurrent(const int* __restrict__ x, const float* __restrict__ emb,
            const bf16* __restrict__ Wi, const bf16* __restrict__ Ws,
            const bf16* __restrict__ Wh, const bf16* __restrict__ Wt,
            const bf16* __restrict__ Wn, const bf16* __restrict__ Wc,
            const bf16* __restrict__ cbT, const float* __restrict__ cb,
            bf16* __restrict__ Hbf, float* __restrict__ stk,
            float* __restrict__ amAcc, float* __restrict__ anAcc,
            float* __restrict__ out) {
  extern __shared__ char smem[];
  bf16*  sA     = (bf16*)(smem);            // 32x256 bf16: emb, then h_nt
  bf16*  sH     = (bf16*)(smem + 16384);    // 32x256 bf16: h / s
  bf16*  sTop   = (bf16*)(smem + 32768);    // 32x128 bf16: popped stack tops
  float* sCodes = (float*)(smem + 40960);   // 64x128 f32 : raw codes
  bf16*  sCodesB= (bf16*) (smem + 73728);   // 64x128 bf16: codes (A operand)
  float* sCbN   = (float*)(smem + 90112);   // 512 f32    : ||codebook||^2
  float* sMinV  = (float*)(smem + 92160);   // 8x64
  int*   sMinI  = (int*)  (smem + 94208);   // 8x64
  int*   sPtr   = (int*)  (smem + 96256);   // 32
  int*   sTok   = (int*)  (smem + 96384);   // 32
  int*   sTopIx = (int*)  (smem + 96512);   // 32
  float* sStep  = (float*)(smem + 96640);   // 32
  int*   sQidx  = (int*)  (smem + 96768);   // 64
  int*   sPos   = (int*)  (smem + 97024);   // 64

  const int tid = threadIdx.x, lane = tid & 31, w = tid >> 5;

  // codebook squared norms (codebook stays L2-resident after first touch)
  for (int n = tid; n < NT_; n += 256) {
    float s = 0.f;
    for (int c = 0; c < CD_; ++c) { float v = cb[n * CD_ + c]; s += v * v; }
    sCbN[n] = s;
  }
  if (tid < B_) sPtr[tid] = 1;                       // bottom sentinel pushed
  for (int i = tid; i < 32 * 256; i += 256) sH[i] = (bf16)0.f;  // h0 = 0
  __syncthreads();

  for (int t = 0; t < L_; ++t) {
    // ---- pop + token gather (thread per batch row) ----
    if (tid < B_) {
      int b = tid;
      int tok = x[b * L_ + t];
      sTok[b] = tok;
      int p = sPtr[b];
      int alive = (p > 0) ? 1 : 0;
      sTopIx[b] = alive ? (p - 1) : -1;
      sStep[b] = ((tok != 0) ? 1.f : 0.f) * (float)alive;
      sPtr[b] = p - alive;
    }
    __syncthreads();
    {
      int b = tid >> 3, c0 = (tid & 7) * 16;
      int ix = sTopIx[b];
      for (int c = 0; c < 16; ++c)
        sTop[b * CD_ + c0 + c] =
            (bf16)((ix >= 0) ? stk[((long)b * DEPTH_ + ix) * CD_ + c0 + c] : 0.f);
    }
    {
      int b = tid >> 3, d0 = (tid & 7) * 32;
      long tok = sTok[tid >> 3];
      for (int d = 0; d < 32; ++d)
        sA[b * D_ + d0 + d] = (bf16)emb[tok * D_ + d0 + d];
    }
    __syncthreads();

    // ---- GEMM1: s = tanh(e@Wi + top@Ws + h@Wh), wave strip n0 = w*32 ----
    {
      const int n0 = w * 32;
      v8f acc[2][2] = {{vzero8(), vzero8()}, {vzero8(), vzero8()}};
      gemm_strip(sA,   D_,  D_,  Wi, n0, lane, acc);
      gemm_strip(sH,   D_,  D_,  Wh, n0, lane, acc);
      gemm_strip(sTop, CD_, CD_, Ws, n0, lane, acc);
      __syncthreads();                        // all reads of sH done
#pragma unroll
      for (int mt = 0; mt < 2; ++mt)
#pragma unroll
        for (int nt = 0; nt < 2; ++nt)
#pragma unroll
          for (int r = 0; r < 8; ++r) {
            int m = mt * 16 + ((lane >> 4) & 1) * 8 + r;
            int n = n0 + nt * 16 + (lane & 15);
            sH[m * D_ + n] = (bf16)tanhf(acc[mt][nt][r]);
          }
      __syncthreads();
    }

    // ---- GEMM2: h_t = tanh(s@Wt) -> Hbf ;  h_nt = tanh(s@Wn) -> sA ----
    {
      const int n0 = w * 32;
      v8f at[2][2] = {{vzero8(), vzero8()}, {vzero8(), vzero8()}};
      v8f an[2][2] = {{vzero8(), vzero8()}, {vzero8(), vzero8()}};
      gemm_strip(sH, D_, D_, Wt, n0, lane, at);
      gemm_strip(sH, D_, D_, Wn, n0, lane, an);
#pragma unroll
      for (int mt = 0; mt < 2; ++mt)
#pragma unroll
        for (int nt = 0; nt < 2; ++nt)
#pragma unroll
          for (int r = 0; r < 8; ++r) {
            int m = mt * 16 + ((lane >> 4) & 1) * 8 + r;
            int n = n0 + nt * 16 + (lane & 15);
            Hbf[((long)t * B_ + m) * D_ + n] = (bf16)tanhf(at[mt][nt][r]);
            sA[m * D_ + n] = (bf16)tanhf(an[mt][nt][r]);
          }
      __syncthreads();
    }

    // ---- GEMM3: codes = h_nt @ W_code  (32x256 -> (b,2,128)) ----
    {
      const int n0 = w * 32;
      v8f ac[2][2] = {{vzero8(), vzero8()}, {vzero8(), vzero8()}};
      gemm_strip(sA, D_, D_, Wc, n0, lane, ac);
#pragma unroll
      for (int mt = 0; mt < 2; ++mt)
#pragma unroll
        for (int nt = 0; nt < 2; ++nt)
#pragma unroll
          for (int r = 0; r < 8; ++r) {
            int b = mt * 16 + ((lane >> 4) & 1) * 8 + r;
            int n = n0 + nt * 16 + (lane & 15);
            int k = n >> 7, cd = n & 127;
            float v = ac[mt][nt][r];
            out[OUT_CODES + (((long)b * L_ + t) * 2 + k) * CD_ + cd] = v;
            sCodes[(b * 2 + k) * CD_ + cd] = v;
            sCodesB[(b * 2 + k) * CD_ + cd] = (bf16)v;
          }
      __syncthreads();
    }

    // ---- quantize: argmin_n ||code - cb_n||^2 via WMMA dot products ----
    {
      float mv[4][8]; int mi[4][8];
#pragma unroll
      for (int mt = 0; mt < 4; ++mt)
#pragma unroll
        for (int r = 0; r < 8; ++r) { mv[mt][r] = 3.4e38f; mi[mt][r] = 0; }

      for (int ntile = 0; ntile < 4; ++ntile) {
        int n0 = w * 64 + ntile * 16;
        v8f sc[4] = {vzero8(), vzero8(), vzero8(), vzero8()};
        for (int k0 = 0; k0 < CD_; k0 += 32) {
          v16bf bb = load_b_swz(cbT, CD_ >> 5, k0, n0, lane);
#pragma unroll
          for (int mt = 0; mt < 4; ++mt) {
            v16bf aa = load_a_ldsbf(sCodesB, CD_, mt * 16, k0, lane);
            sc[mt] = wmma_bf16(aa, bb, sc[mt]);
          }
        }
        int n = n0 + (lane & 15);
        float cbn = sCbN[n];
#pragma unroll
        for (int mt = 0; mt < 4; ++mt)
#pragma unroll
          for (int r = 0; r < 8; ++r) {
            float d = cbn - 2.f * sc[mt][r];
            if (d < mv[mt][r] || (d == mv[mt][r] && n < mi[mt][r])) {
              mv[mt][r] = d; mi[mt][r] = n;
            }
          }
      }
      // reduce across the 16 lanes that hold one C-row (xor stays in-group)
#pragma unroll
      for (int off = 1; off < 16; off <<= 1)
#pragma unroll
        for (int mt = 0; mt < 4; ++mt)
#pragma unroll
          for (int r = 0; r < 8; ++r) {
            float ov = __shfl_xor(mv[mt][r], off, 32);
            int   oi = __shfl_xor(mi[mt][r], off, 32);
            if (ov < mv[mt][r] || (ov == mv[mt][r] && oi < mi[mt][r])) {
              mv[mt][r] = ov; mi[mt][r] = oi;
            }
          }
      if ((lane & 15) == 0) {
        int hi = lane >> 4;
#pragma unroll
        for (int mt = 0; mt < 4; ++mt)
#pragma unroll
          for (int r = 0; r < 8; ++r) {
            int row = mt * 16 + hi * 8 + r;
            sMinV[w * 64 + row] = mv[mt][r];
            sMinI[w * 64 + row] = mi[mt][r];
          }
      }
    }
    __syncthreads();
    if (tid < 64) {
      float bv = sMinV[tid]; int bi = sMinI[tid];
      for (int ww = 1; ww < 8; ++ww) {
        float v = sMinV[ww * 64 + tid]; int ii = sMinI[ww * 64 + tid];
        if (v < bv || (v == bv && ii < bi)) { bv = v; bi = ii; }
      }
      sQidx[tid] = bi;
      int b = tid >> 1, k = tid & 1;
      out[OUT_PUSH + (((long)b * L_ + t) * 2 + k)] = (float)bi * sStep[b];
    }
    __syncthreads();

    // ---- push positions (sequential per batch row) ----
    if (tid < B_) {
      int b = tid, p = sPtr[b];
      float sm = sStep[b];
#pragma unroll
      for (int k = 0; k < 2; ++k) {
        int q = sQidx[b * 2 + k];
        int doPush = (q != 0 && sm != 0.f) ? 1 : 0;
        sPos[b * 2 + k] = doPush ? p : -1;
        p += doPush;
      }
      sPtr[b] = p;
    }
    __syncthreads();
    // cooperative stack push of quantized codebook rows
    {
      int row = tid >> 2, c0 = (tid & 3) * 32;
      int pos = sPos[row];
      if (pos >= 0) {
        int b = row >> 1; long q = sQidx[row];
        for (int c = 0; c < 32; ++c)
          stk[((long)b * DEPTH_ + pos) * CD_ + c0 + c] = cb[q * CD_ + c0 + c];
      }
    }
    // ---- EMA accumulators (includes qidx==0 per reference one-hot) ----
    if (tid < 64) {
      float sm = sStep[tid >> 1];
      if (sm != 0.f) atomicAdd(anAcc + sQidx[tid], sm);
    }
    {
      int row = tid >> 2, c0 = (tid & 3) * 32;
      float sm = sStep[row >> 1];
      if (sm != 0.f) {
        long q = sQidx[row];
        for (int c = 0; c < 32; ++c)
          atomicAdd(amAcc + q * CD_ + c0 + c, sm * sCodes[row * CD_ + c0 + c]);
      }
    }
    __threadfence();   // stack writes must be visible to next step's pop
    __syncthreads();
  }
}

// ---------------------------------------------------------------------------
// K3: logits = H(4096x256,bf16) @ Wout(swizzled bf16) + b_out, scattered to
// (b, t, V).  HBM-store-bound: 524 MB f32 out => ~22.5 us floor @ 23.3 TB/s.
// The block's full 256x128 B-tile (64 KB) is staged in LDS ONCE (pure b128
// memcpy in fragment layout) so the 8 waves share it instead of each pulling
// it from L2 (8x redundancy eliminated); per-wave reads are ds_load_b128.
// ---------------------------------------------------------------------------
__global__ void __launch_bounds__(256)
k_logits(const bf16* __restrict__ Hbf, const bf16* __restrict__ Wsz,
         const float* __restrict__ bo, float* __restrict__ out) {
  extern __shared__ char smemL[];
  bf16* sB = (bf16*)smemL;            // [kc][nt][512] = 32768 bf16 = 64 KB
  const int tid = threadIdx.x, lane = tid & 31, w = tid >> 5;
  const long M0 = (long)blockIdx.y * 128 + w * 16;
  const int  N0 = blockIdx.x * 128;
  const int  kfr = D_ >> 5;           // 8 k-fragments per column strip

  // stage whole B tile: 4096 16-byte units, 16 per thread, all coalesced
#pragma unroll
  for (int j = 0; j < 16; ++j) {
    int u    = tid + j * 256;         // unit index
    int e8   = u & 1;
    int ulan = (u >> 1) & 31;
    int nt   = (u >> 6) & 7;
    int kc   = u >> 9;
    const bf16* src = Wsz + ((long)((N0 >> 4) + nt) * kfr + kc) * 512
                          + ulan * 16 + e8 * 8;
    *(v8bf*)(sB + u * 8) = *(const v8bf*)(src);
  }
  __syncthreads();

  v8f acc[8];
#pragma unroll
  for (int nt = 0; nt < 8; ++nt) acc[nt] = vzero8();

#pragma unroll
  for (int kc = 0; kc < 8; ++kc) {
    int k0 = kc * 32;
    if (kc + 1 < 8)  // cover next A slab (gfx1250 global_prefetch_b8)
      __builtin_prefetch(Hbf + (M0 + (lane & 15)) * D_ + k0 + 32, 0, 1);
    v16bf a = load_a_gbf(Hbf, D_, M0, k0, lane);
#pragma unroll
    for (int nt = 0; nt < 8; ++nt) {
      const bf16* p = sB + (kc * 8 + nt) * 512 + lane * 16;
      v16bf bb = cat8(*(const v8bf*)(p), *(const v8bf*)(p + 8));
      acc[nt] = wmma_bf16(a, bb, acc[nt]);
    }
  }
#pragma unroll
  for (int nt = 0; nt < 8; ++nt)
#pragma unroll
    for (int r = 0; r < 8; ++r) {
      long m = M0 + ((lane >> 4) & 1) * 8 + r;     // row = t*32 + b
      int  n = N0 + nt * 16 + (lane & 15);
      long t = m >> 5, b = m & 31;
      out[OUT_LOGITS + (b * L_ + t) * (long)V_ + n] = acc[nt][r] + bo[n];
    }
}

// K4: EMA finalize
__global__ void k_finalize(const float* __restrict__ m_acc,
                           const float* __restrict__ n_acc,
                           const float* __restrict__ amAcc,
                           const float* __restrict__ anAcc,
                           float* __restrict__ out) {
  long i = (long)blockIdx.x * blockDim.x + threadIdx.x;
  const float r = 0.99f, ir = 1.f - r;
  if (i < NT_ * CD_) out[OUT_M + i] = r * m_acc[i] + ir * amAcc[i];
  else if (i < NT_ * CD_ + NT_) {
    long j = i - NT_ * CD_;
    out[OUT_N + j] = r * n_acc[j] + ir * anAcc[j];
  }
}

// ---------------------------------------------------------------------------
extern "C" void kernel_launch(void* const* d_in, const int* in_sizes, int n_in,
                              void* d_out, int out_size, void* d_ws, size_t ws_size,
                              hipStream_t stream) {
  const int*   x     = (const int*)  d_in[0];
  const float* emb   = (const float*)d_in[1];
  const float* W_i   = (const float*)d_in[2];
  const float* W_s   = (const float*)d_in[3];
  const float* W_h   = (const float*)d_in[4];
  const float* W_t   = (const float*)d_in[5];
  const float* W_n   = (const float*)d_in[6];
  const float* W_c   = (const float*)d_in[7];
  const float* W_out = (const float*)d_in[8];
  const float* b_out = (const float*)d_in[9];
  const float* cb    = (const float*)d_in[10];
  const float* m_acc = (const float*)d_in[11];
  const float* n_acc = (const float*)d_in[12];
  float* out = (float*)d_out;
  char*  ws  = (char*)d_ws;

  bf16*  Hbf   = (bf16*)(ws + WS_HBF);
  bf16*  Wouts = (bf16*)(ws + WS_WOUTS);
  bf16*  Wis   = (bf16*)(ws + WS_WI);
  bf16*  Whs   = (bf16*)(ws + WS_WH);
  bf16*  Wss   = (bf16*)(ws + WS_WS);
  bf16*  Wts   = (bf16*)(ws + WS_WT);
  bf16*  Wns   = (bf16*)(ws + WS_WN);
  bf16*  Wcs   = (bf16*)(ws + WS_WC);
  bf16*  cbT   = (bf16*)(ws + WS_CBT);
  float* amAcc = (float*)(ws + WS_AM);
  float* anAcc = (float*)(ws + WS_AN);
  float* stk   = (float*)(ws + WS_STACK);

  {
    long n = (long)NT_ * CD_ + NT_ + (long)B_ * DEPTH_ * CD_;
    k_zero_ws<<<(int)((n + 255) / 256), 256, 0, stream>>>(amAcc, anAcc, stk);
  }
  // weight swizzles (f32 -> fragment-native bf16), done once per call
  {
    long n = (long)D_ * V_;     // W_out
    k_swizzle<<<(int)((n + 255) / 256), 256, 0, stream>>>(W_out, Wouts, D_, V_, V_, 0);
    long m = (long)D_ * D_;     // 256x256 weights
    int  gb = (int)((m + 255) / 256);
    k_swizzle<<<gb, 256, 0, stream>>>(W_i, Wis, D_, D_, D_, 0);
    k_swizzle<<<gb, 256, 0, stream>>>(W_h, Whs, D_, D_, D_, 0);
    k_swizzle<<<gb, 256, 0, stream>>>(W_t, Wts, D_, D_, D_, 0);
    k_swizzle<<<gb, 256, 0, stream>>>(W_n, Wns, D_, D_, D_, 0);
    k_swizzle<<<gb, 256, 0, stream>>>(W_c, Wcs, D_, 2 * CD_, 2 * CD_, 0);
    long s = (long)CD_ * D_;    // W_s: K=128, N=256
    k_swizzle<<<(int)((s + 255) / 256), 256, 0, stream>>>(W_s, Wss, CD_, D_, D_, 0);
    long c = (long)CD_ * NT_;   // codebook^T: K=128, N=512, src transposed
    k_swizzle<<<(int)((c + 255) / 256), 256, 0, stream>>>(cb, cbT, CD_, NT_, CD_, 1);
  }
  k_recurrent<<<1, 256, 97280, stream>>>(x, emb, Wis, Wss, Whs, Wts, Wns, Wcs,
                                         cbT, cb, Hbf, stk, amAcc, anAcc, out);
  {
    dim3 grid(V_ / 128, (B_ * L_) / 128);   // 250 x 32
    k_logits<<<grid, 256, 65536, stream>>>(Hbf, Wouts, b_out, out);
  }
  {
    long n = (long)NT_ * CD_ + NT_;
    k_finalize<<<(int)((n + 255) / 256), 256, 0, stream>>>(m_acc, n_acc,
                                                           amAcc, anAcc, out);
  }
}
